// Offset2D_11544872092059
// MI455X (gfx1250) — matrix operand order repaired and checked by reading, not compiled, not run on hardware
//
#include <hip/hip_runtime.h>

typedef __attribute__((ext_vector_type(16))) _Float16 v16h;
typedef __attribute__((ext_vector_type(8)))  float    v8f;

#define B_   8
#define C_   64
#define H_   256
#define W_   256
#define HW_  (H_*W_)         // 65536
#define DH_  128
#define DW_  128
#define DHW_ (DH_*DW_)       // 16384
#define CP_  65
#define EPS_ 1e-5f

// output layout (flat f32, reference return order)
#define DOWN_ELEMS (B_*CP_*DHW_)            // 8,519,680
#define OFF_BASE   DOWN_ELEMS               // offset: (B,2,H,W)
#define ATT_BASE   (OFF_BASE + B_*2*HW_)    // attention: (B,1,H,W)
#define DEST_BASE  (ATT_BASE + B_*HW_)      // destination: (B,2,H,W) stored as f32

// ---------------------------------------------------------------- zero `down`
__global__ void zero_down_kernel(float* __restrict__ out) {
    float4 z = {0.f, 0.f, 0.f, 0.f};
    int stride = gridDim.x * blockDim.x;
    for (int i = blockIdx.x * blockDim.x + threadIdx.x; i < DOWN_ELEMS / 4; i += stride)
        ((float4*)out)[i] = z;
}

// -------------------------------------------------- fused conv+index+scatter
__global__ __launch_bounds__(256) void offset2d_fused_kernel(
        const float* __restrict__ x,      // (B,64,H,W)
        const float* __restrict__ cw,     // (3,64)
        const float* __restrict__ cb,     // (3,)
        float* __restrict__ out) {
    const int lane  = threadIdx.x & 31;
    const int m     = lane & 15;                      // row-in-tile / pixel-in-half
    const int wave  = blockIdx.x * (blockDim.x >> 5) + (threadIdx.x >> 5);
    const int nwav  = gridDim.x * (blockDim.x >> 5);

    // ---- A operand: weights, M=16 (3 used), K=64 in two 16x32 f16 chunks.
    // A layout (16-bit A 16x32): lanes 0-15 hold K={0..7,16..23}, lanes 16-31 K={8..15,24..31}
    const int ahalf = (lane >> 4) ? 8 : 0;
    v16h a[2];
    for (int q = 0; q < 2; ++q)
        for (int j = 0; j < 8; ++j) {
            int kb = ((j < 4) ? (2 * j) : (16 + 2 * (j - 4))) + ahalf + 32 * q;
            _Float16 w0 = (_Float16)0.f, w1 = (_Float16)0.f;
            if (m < 3) {                       // pad M=3 -> 16 with zeros
                w0 = (_Float16)cw[m * 64 + kb];
                w1 = (_Float16)cw[m * 64 + kb + 1];
            }
            a[q][2 * j]     = w0;
            a[q][2 * j + 1] = w1;
        }
    const float b0 = cb[0], b1 = cb[1], b2 = cb[2];

    // B layout (16-bit B 32x16): lanes 0-15 hold K={0..7,16..23}, lanes 16-31 K={8..15,24..31}
    const int bhalf = (lane >> 4) ? 8 : 0;

    const int total_chunks = B_ * HW_ / 32;           // 16384 chunks of 32 pixels
    for (int chunk = wave; chunk < total_chunks; chunk += nwav) {
        const int b       = chunk / (HW_ / 32);
        const int pixbase = (chunk % (HW_ / 32)) * 32; // 32 consecutive w (W=256)
        const float* xb   = x + (size_t)b * C_ * HW_;

        // ---- 1x1 conv via WMMA: two 16-pixel tiles, K=64 in two steps
        v8f acc0 = {}, acc1 = {};
        for (int q = 0; q < 2; ++q) {
            v16h bt0, bt1;
            const int p0 = pixbase + m;
            const int p1 = pixbase + 16 + m;
            for (int j = 0; j < 8; ++j) {
                int k0 = ((j < 4) ? (2 * j) : (16 + 2 * (j - 4))) + bhalf + 32 * q;
                bt0[2 * j]     = (_Float16)xb[(size_t)k0       * HW_ + p0];
                bt0[2 * j + 1] = (_Float16)xb[(size_t)(k0 + 1) * HW_ + p0];
                bt1[2 * j]     = (_Float16)xb[(size_t)k0       * HW_ + p1];
                bt1[2 * j + 1] = (_Float16)xb[(size_t)(k0 + 1) * HW_ + p1];
            }
            acc0 = __builtin_amdgcn_wmma_f32_16x16x32_f16(false, a[q], false, bt0,
                                                          (short)0, acc0, false, false);
            acc1 = __builtin_amdgcn_wmma_f32_16x16x32_f16(false, a[q], false, bt1,
                                                          (short)0, acc1, false, false);
        }

        // ---- redistribute so lane L owns pixel pixbase+L  (D: M=o lives in
        // VGPR o of lanes 0-15; tile1 pixel L>=16 sits at tile1 lane L-16)
        float t00 = __shfl(acc0[0], m, 32), t10 = __shfl(acc1[0], m, 32);
        float t01 = __shfl(acc0[1], m, 32), t11 = __shfl(acc1[1], m, 32);
        float t02 = __shfl(acc0[2], m, 32), t12 = __shfl(acc1[2], m, 32);
        float oa0 = ((lane < 16) ? t00 : t10) + b0;   // offset_y
        float oa1 = ((lane < 16) ? t01 : t11) + b1;   // offset_x
        float oa2 = ((lane < 16) ? t02 : t12) + b2;   // attention

        const int pix  = pixbase + lane;              // index within (H,W)
        const int h    = pix >> 8;
        const int wcol = pix & 255;

        // ---- destination grid
        float gy    = (float)h    * (1.0f / 256.0f);
        float gx    = (float)wcol * (1.0f / 256.0f);
        float desty = fminf(fmaxf(gy + oa0, 0.0f), 1.0f - EPS_);
        float destx = fminf(fmaxf(gx + oa1, 0.0f), 1.0f - EPS_);
        int   dy    = (int)(desty * (float)DH_);      // values >= 0: trunc == floor
        int   dx    = (int)(destx * (float)DW_);

        // ---- dense per-pixel outputs (coalesced: lanes = consecutive w)
        out[OFF_BASE  + ((size_t)(b * 2 + 0)) * HW_ + pix] = oa0;
        out[OFF_BASE  + ((size_t)(b * 2 + 1)) * HW_ + pix] = oa1;
        out[ATT_BASE  + (size_t)b * HW_ + pix]             = oa2;
        out[DEST_BASE + ((size_t)(b * 2 + 0)) * HW_ + pix] = (float)dy;
        out[DEST_BASE + ((size_t)(b * 2 + 1)) * HW_ + pix] = (float)dx;

        // ---- scatter-add 64 channels + ones/65536 (fold the final divide).
        // x re-reads hit L0/L2 (same 8KB just read for WMMA). down (34MB) is
        // L2-resident, so these f32 atomics resolve in L2 at full rate.
        const int bin = dy * DW_ + dx;
        float* downb  = out + (size_t)b * CP_ * DHW_;
        #pragma unroll 4
        for (int c = 0; c < C_; ++c) {
            float v = xb[(size_t)c * HW_ + pix];
            unsafeAtomicAdd(downb + (size_t)c * DHW_ + bin, v);
        }
        unsafeAtomicAdd(downb + (size_t)64 * DHW_ + bin, 1.0f / 65536.0f);
    }
}

extern "C" void kernel_launch(void* const* d_in, const int* in_sizes, int n_in,
                              void* d_out, int out_size, void* d_ws, size_t ws_size,
                              hipStream_t stream) {
    const float* x  = (const float*)d_in[0];   // (8,64,256,256) f32
    const float* cw = (const float*)d_in[1];   // (3,64) f32
    const float* cb = (const float*)d_in[2];   // (3,) f32
    float* out = (float*)d_out;

    // down region must be zeroed every call (harness does not re-poison)
    zero_down_kernel<<<1024, 256, 0, stream>>>(out);

    // 512 blocks x 8 waves = 4096 waves; 16384 chunks -> uniform 4 iters/wave,
    // EXEC stays all-ones at every WMMA.
    offset2d_fused_kernel<<<512, 256, 0, stream>>>(x, cw, cb, out);
}
